// RGCNModel_28982439313534
// MI455X (gfx1250) — compile-verified, matrix operand order checked by reading, not run on hardware
//
#include <hip/hip_runtime.h>
#include <hip/hip_bf16.h>

// ---------------------------------------------------------------------------
// Problem constants (match reference)
// ---------------------------------------------------------------------------
#define NREL   16
#define NCOLS  1088   // 16 relations * 64 + 64 root columns
#define NCTILE 17     // 1088 / 64

typedef __attribute__((ext_vector_type(16))) __bf16 v16bf;
typedef __attribute__((ext_vector_type(8)))  float  v8f;

// ---------------------------------------------------------------------------
// Zero a float buffer (grid-stride).
// ---------------------------------------------------------------------------
__global__ void rgcn_zero_f32(float* __restrict__ p, size_t n) {
    size_t i = (size_t)blockIdx.x * blockDim.x + threadIdx.x;
    size_t stride = (size_t)gridDim.x * blockDim.x;
    for (; i < n; i += stride) p[i] = 0.0f;
}

// ---------------------------------------------------------------------------
// Per-(dst, rel) in-degree via f32 atomics (deg buffer is 6.4 MB -> L2 hot).
// ---------------------------------------------------------------------------
__global__ void rgcn_degree(const int* __restrict__ edge_index,
                            const int* __restrict__ edge_type,
                            float* __restrict__ deg, int E) {
    int e = blockIdx.x * blockDim.x + threadIdx.x;
    if (e >= E) return;
    int dst = edge_index[E + e];
    int r   = edge_type[e];
    atomicAdd(&deg[(size_t)dst * NREL + r], 1.0f);
}

// ---------------------------------------------------------------------------
// Pre-pack Bcat = [ W[0] | ... | W[15] | root ]  ([K,1088] logical, f32)
// into bf16 WMMA B fragments, fragment-major:
//   frag = (ctile*ksteps + kstep)*4 + cc ;  32 lanes x 16 bf16 (32 B) each.
// Per CDNA5 16-bit 32x16 B layout:
//   lanes 0-15 : col N=l,    K = {0..7} in v0-3, {16..23} in v4-7
//   lanes 16-31: col N=l-16, K = {8..15} in v0-3, {24..31} in v4-7
// One-shot kernel (<5K threads): all branchy gather cost lives here, not in
// the GEMM hot loop.
// ---------------------------------------------------------------------------
__global__ void rgcn_pack_b(const float* __restrict__ W,     // [NREL, K, 64]
                            const float* __restrict__ root,  // [K, 64]
                            __bf16* __restrict__ bp, int K) {
    const int ksteps = K >> 5;
    const int total  = NCTILE * ksteps * 4 * 32;
    int t = blockIdx.x * blockDim.x + threadIdx.x;
    if (t >= total) return;
    const int lane  = t & 31;
    const int frag  = t >> 5;
    const int cc    = frag & 3;
    const int kstep = (frag >> 2) % ksteps;
    const int ctile = (frag >> 2) / ksteps;

    const int lmod  = lane & 15;
    const int khalf = (lane < 16) ? 0 : 8;
    const int col   = ctile * 64 + cc * 16 + lmod;

    __bf16* outp = bp + (size_t)t * 16;         // 32 B per lane, contiguous
#pragma unroll
    for (int j = 0; j < 8; ++j) {
        int kb = (kstep << 5) + ((j >> 2) << 4) + khalf + ((j & 3) << 1);
        float e0, e1;
        if (col < NREL * 64) {                  // relation weight column
            int r = col >> 6, o = col & 63;
            const float* wb = W + ((size_t)r * K + kb) * 64 + o;
            e0 = wb[0];
            e1 = wb[64];                        // (k+1) row, stride 64 floats
        } else {                                // root column
            int o = col - NREL * 64;
            e0 = root[(size_t)kb * 64 + o];
            e1 = root[(size_t)(kb + 1) * 64 + o];
        }
        outp[2 * j]     = (__bf16)e0;
        outp[2 * j + 1] = (__bf16)e1;
    }
}

// ---------------------------------------------------------------------------
// WMMA GEMM:  out[N, 1088] = bf16(X[N,K]) @ Bpack
// One wave -> one 16(M) x 64(cols) output tile, f32 accumulate.
// A fragment: two aligned contiguous 32-B f32 loads per lane + packed
// bf16 converts. B fragment: one aligned 32-B load from pre-packed buffer
// (static immediate offsets, coalesced, L0/L2-resident).
// Store epilogue: unguarded fast path for full tiles (all but the last
// M tile), guarded path only for the ragged tail.
// ---------------------------------------------------------------------------
__global__ void rgcn_gemm_wmma(const float*  __restrict__ X,     // [Nrows, K]
                               const __bf16* __restrict__ Bpack, // packed frags
                               float* __restrict__ out,          // [Nrows, 1088]
                               int Nrows, int K) {
    const int lane   = threadIdx.x & 31;
    const int wib    = threadIdx.x >> 5;
    const int wpb    = blockDim.x >> 5;
    const int gwave  = blockIdx.x * wpb + wib;
    const int mtile  = gwave / NCTILE;
    const int ctile  = gwave % NCTILE;
    if (mtile * 16 >= Nrows) return;            // wave-uniform exit

    const int ksteps = K >> 5;
    const int m0     = mtile * 16;
    const int c0     = ctile * 64;
    const int lmod   = lane & 15;
    const int khalf  = (lane < 16) ? 0 : 8;

    int rowA = m0 + lmod;
    if (rowA >= Nrows) rowA = Nrows - 1;        // clamp: stores guarded in tail

    v8f acc[4] = {};                            // 4 x (16x16 f32) = 16x64 tile

    for (int kstep = 0; kstep < ksteps; ++kstep) {
        // ---- A fragment: two contiguous 32-B runs of the X row ----
        const float* xrow = X + (size_t)rowA * K + (kstep << 5);
        v8f lo = *(const v8f*)(xrow + khalf);
        v8f hi = *(const v8f*)(xrow + 16 + khalf);
        v16bf a;
#pragma unroll
        for (int i = 0; i < 8; ++i) {
            a[i]     = (__bf16)lo[i];
            a[8 + i] = (__bf16)hi[i];
        }
        // ---- 4 column sub-tiles: packed B fragment = one 32-B load ----
        const size_t fbase = ((size_t)(ctile * ksteps + kstep) * 4);
#pragma unroll
        for (int cc = 0; cc < 4; ++cc) {
            const v16bf b =
                *(const v16bf*)(Bpack + ((fbase + cc) * 32 + lane) * 16);
            acc[cc] = __builtin_amdgcn_wmma_f32_16x16x32_bf16(
                false, a, false, b, (short)0, acc[cc], false, false);
        }
    }

    // ---- store 16x64 f32 tile (D layout: lane%16 = col, vgpr v = row) ----
    const int mrow0 = m0 + ((lane < 16) ? 0 : 8);
    float* obase = out + (size_t)mrow0 * NCOLS + c0 + lmod;
    if (m0 + 16 <= Nrows) {
        // Fast path (6249/6250 tiles): no guards, stores coalesce + clause.
#pragma unroll
        for (int cc = 0; cc < 4; ++cc)
#pragma unroll
            for (int v = 0; v < 8; ++v)
                obase[(size_t)v * NCOLS + cc * 16] = acc[cc][v];
    } else {
        // Ragged tail tile only.
#pragma unroll
        for (int cc = 0; cc < 4; ++cc)
#pragma unroll
            for (int v = 0; v < 8; ++v)
                if (mrow0 + v < Nrows)
                    obase[(size_t)v * NCOLS + cc * 16] = acc[cc][v];
    }
}

// ---------------------------------------------------------------------------
// Wave-per-edge scatter: agg[dst,:] += xw[src, etype*64 : +64] / c_{dst,r}
// 64 consecutive floats per edge -> coalesced loads + atomics; agg is
// 25.6 MB -> resident in 192 MB L2, atomic RMWs stay on-chip.
// ---------------------------------------------------------------------------
__global__ void rgcn_scatter(const int* __restrict__ edge_index,
                             const int* __restrict__ edge_type,
                             const float* __restrict__ deg,
                             const float* __restrict__ xw,   // [N, 1088]
                             float* __restrict__ agg,        // [N, 64]
                             int E) {
    int wid  = (blockIdx.x * blockDim.x + threadIdx.x) >> 5;
    int lane = threadIdx.x & 31;
    if (wid >= E) return;
    int src = edge_index[wid];
    int dst = edge_index[E + wid];
    int r   = edge_type[wid];
    float d    = deg[(size_t)dst * NREL + r];
    float norm = 1.0f / fmaxf(d, 1.0f);
    const float* row  = xw  + (size_t)src * NCOLS + (size_t)r * 64;
    float*       arow = agg + (size_t)dst * 64;
    __builtin_prefetch(row, 0, 1);              // global_prefetch_b8
    atomicAdd(&arow[lane],      row[lane]      * norm);
    atomicAdd(&arow[lane + 32], row[lane + 32] * norm);
}

// ---------------------------------------------------------------------------
// h[n,o] = relu(agg[n,o] + xw[n, 1024+o] + b[o])   (root part rides in xw)
// ---------------------------------------------------------------------------
__global__ void rgcn_node_update(const float* __restrict__ agg,
                                 const float* __restrict__ xw,
                                 const float* __restrict__ bias,
                                 float* __restrict__ h, int Nn) {
    int idx = blockIdx.x * blockDim.x + threadIdx.x;
    if (idx >= Nn * 64) return;
    int n = idx >> 6, o = idx & 63;
    float v = agg[idx] + xw[(size_t)n * NCOLS + NREL * 64 + o] + bias[o];
    h[idx] = fmaxf(v, 0.0f);
}

// ---------------------------------------------------------------------------
// DistMult: score[e] = sum_d h2[src,d] * rel[etype,d] * h2[dst,d]
// Wave-per-edge, wave32 xor-shuffle tree reduction.
// ---------------------------------------------------------------------------
__global__ void rgcn_score(const int* __restrict__ edge_index,
                           const int* __restrict__ edge_type,
                           const float* __restrict__ h2,
                           const float* __restrict__ rel,
                           float* __restrict__ out, int E) {
    int wid  = (blockIdx.x * blockDim.x + threadIdx.x) >> 5;
    int lane = threadIdx.x & 31;
    if (wid >= E) return;
    int src = edge_index[wid];
    int dst = edge_index[E + wid];
    int r   = edge_type[wid];
    const float* hs = h2  + (size_t)src * 64;
    const float* hd = h2  + (size_t)dst * 64;
    const float* rr = rel + (size_t)r * 64;
    float p = hs[lane] * rr[lane] * hd[lane]
            + hs[lane + 32] * rr[lane + 32] * hd[lane + 32];
#pragma unroll
    for (int off = 16; off; off >>= 1) p += __shfl_xor(p, off, 32);
    if (lane == 0) out[wid] = p;
}

// ---------------------------------------------------------------------------
// Launch
// ---------------------------------------------------------------------------
extern "C" void kernel_launch(void* const* d_in, const int* in_sizes, int n_in,
                              void* d_out, int out_size, void* d_ws, size_t ws_size,
                              hipStream_t stream) {
    const int*   edge_index = (const int*)d_in[0];   // [2, E]
    const int*   edge_type  = (const int*)d_in[1];   // [E]
    const float* init_emb   = (const float*)d_in[2]; // [N, 32]
    const float* W1         = (const float*)d_in[3]; // [16, 32, 64]
    const float* root1      = (const float*)d_in[4]; // [32, 64]
    const float* b1         = (const float*)d_in[5]; // [64]
    const float* W2         = (const float*)d_in[6]; // [16, 64, 64]
    const float* root2      = (const float*)d_in[7]; // [64, 64]
    const float* b2         = (const float*)d_in[8]; // [64]
    const float* rel_emb    = (const float*)d_in[9]; // [16, 64]
    float*       score      = (float*)d_out;         // [E]

    const int E = in_sizes[1];
    const int N = in_sizes[2] / 32;

    // Workspace carve-up (xw/agg/Bpack reused across layers): ~518 MB total.
    char* ws = (char*)d_ws;
    float*  deg   = (float*)ws;                      ws += (size_t)N * NREL * 4;
    float*  xw    = (float*)ws;                      ws += (size_t)N * NCOLS * 4;
    float*  agg   = (float*)ws;                      ws += (size_t)N * 64 * 4;
    float*  h1    = (float*)ws;                      ws += (size_t)N * 64 * 4;
    float*  h2    = (float*)ws;                      ws += (size_t)N * 64 * 4;
    __bf16* bpack = (__bf16*)ws;                     ws += 256 * 1024; // <=140KB used
    (void)ws_size; (void)n_in;

    const int mtiles     = (N + 15) / 16;
    const int gemmWaves  = mtiles * NCTILE;
    const int gemmBlocks = (gemmWaves + 7) / 8;      // 8 waves / 256-thr block
    const int edgeWaveBlocks = (E + 7) / 8;          // wave-per-edge kernels
    const int updBlocks  = (N * 64 + 255) / 256;
    const int pack1Thr   = NCTILE * (32 / 32) * 4 * 32;
    const int pack2Thr   = NCTILE * (64 / 32) * 4 * 32;

    // ---- normalization degrees ----
    rgcn_zero_f32<<<2048, 256, 0, stream>>>(deg, (size_t)N * NREL);
    rgcn_degree<<<(E + 255) / 256, 256, 0, stream>>>(edge_index, edge_type, deg, E);

    // ---- layer 1 ----
    rgcn_pack_b<<<(pack1Thr + 255) / 256, 256, 0, stream>>>(W1, root1, bpack, 32);
    rgcn_gemm_wmma<<<gemmBlocks, 256, 0, stream>>>(init_emb, bpack, xw, N, 32);
    rgcn_zero_f32<<<2048, 256, 0, stream>>>(agg, (size_t)N * 64);
    rgcn_scatter<<<edgeWaveBlocks, 256, 0, stream>>>(edge_index, edge_type, deg, xw, agg, E);
    rgcn_node_update<<<updBlocks, 256, 0, stream>>>(agg, xw, b1, h1, N);

    // ---- layer 2 (reuse xw / agg / bpack buffers) ----
    rgcn_pack_b<<<(pack2Thr + 255) / 256, 256, 0, stream>>>(W2, root2, bpack, 64);
    rgcn_gemm_wmma<<<gemmBlocks, 256, 0, stream>>>(h1, bpack, xw, N, 64);
    rgcn_zero_f32<<<2048, 256, 0, stream>>>(agg, (size_t)N * 64);
    rgcn_scatter<<<edgeWaveBlocks, 256, 0, stream>>>(edge_index, edge_type, deg, xw, agg, E);
    rgcn_node_update<<<updBlocks, 256, 0, stream>>>(agg, xw, b2, h2, N);

    // ---- DistMult scoring ----
    rgcn_score<<<edgeWaveBlocks, 256, 0, stream>>>(edge_index, edge_type, h2, rel_emb, score, E);
    (void)out_size;
}
